// BasicBlock_44693429682673
// MI455X (gfx1250) — matrix operand order repaired and checked by reading, not compiled, not run on hardware
//
#include <hip/hip_runtime.h>
#include <hip/hip_bf16.h>

// ---------------------------------------------------------------------------
// Transformer basic block for MI455X (gfx1250, wave32, WMMA + async LDS copy).
// B=2, L=2048, D=1024, H=16, C=64.
// ---------------------------------------------------------------------------

typedef __attribute__((ext_vector_type(8)))  _Float16 v8h;
typedef __attribute__((ext_vector_type(16))) _Float16 v16h;
typedef __attribute__((ext_vector_type(8)))  float    v8f;
typedef __attribute__((ext_vector_type(4)))  int      v4i;

// Explicit LDS (addrspace 3) types so LDS traffic lowers to ds_* (not flat_*).
typedef __attribute__((address_space(3))) _Float16 ldsh;
typedef __attribute__((address_space(3))) v8h      ldsv8h;

#define B_  2
#define L_  2048
#define D_  1024
#define H_  16
#define C_  64
#define SCALE_ 0.03125f   // 1/sqrt(1024)

static __device__ __forceinline__ v8f wmma_f16(v16h a, v16h b, v8f c) {
  return __builtin_amdgcn_wmma_f32_16x16x32_f16(false, a, false, b, (short)0, c,
                                                false, false);
}

// Build a 16-half fragment from two 16-byte global chunks.
static __device__ __forceinline__ v16h ld2(const _Float16* p0, const _Float16* p1) {
  v8h lo = *(const v8h*)p0;
  v8h hi = *(const v8h*)p1;
  v16h r;
#pragma unroll
  for (int i = 0; i < 8; ++i) { r[i] = lo[i]; r[i + 8] = hi[i]; }
  return r;
}
static __device__ __forceinline__ v16h ld16c(const _Float16* p) { return ld2(p, p + 8); }

// Same, from LDS (ds_load_b128 path).
static __device__ __forceinline__ v16h ld2_l(const ldsh* p0, const ldsh* p1) {
  v8h lo = *(const ldsv8h*)p0;
  v8h hi = *(const ldsv8h*)p1;
  v16h r;
#pragma unroll
  for (int i = 0; i < 8; ++i) { r[i] = lo[i]; r[i + 8] = hi[i]; }
  return r;
}
static __device__ __forceinline__ v16h ld16c_l(const ldsh* p) { return ld2_l(p, p + 8); }

// ---------------------------------------------------------------------------
// CDNA5 async global->LDS copy (ASYNCcnt-tracked). Fallback: sync LDS store.
// Builtin params: v4i __device__* (AS1), v4i __shared__* (AS3), imm offset, imm cpol.
// ---------------------------------------------------------------------------
#if defined(__AMDGCN__) && __has_builtin(__builtin_amdgcn_global_load_async_to_lds_b128)
#define HAVE_ASYNC_LDS 1
typedef __attribute__((address_space(1))) v4i* gv4i_p;
typedef __attribute__((address_space(3))) v4i* lv4i_p;
static __device__ __forceinline__ void cp16(const _Float16* g, ldsh* l) {
  __builtin_amdgcn_global_load_async_to_lds_b128((gv4i_p)g, (lv4i_p)l, 0, 0);
}
#else
#define HAVE_ASYNC_LDS 0
static __device__ __forceinline__ void cp16(const _Float16* g, ldsh* l) {
  *(ldsv8h*)l = *(const v8h*)g;
}
#endif

static __device__ __forceinline__ void wait_async_le4() {
#if HAVE_ASYNC_LDS
  asm volatile("s_wait_asynccnt 0x4" ::: "memory");
#endif
}
static __device__ __forceinline__ void wait_async_0() {
#if HAVE_ASYNC_LDS
  asm volatile("s_wait_asynccnt 0x0" ::: "memory");
#endif
}

// Stage A[64x32] (row stride K) and W[64x32] tiles into LDS (row stride 32).
// 128 threads x 2 iterations x (A+B) 16B chunks = 8KB; 4 async ops per wave.
static __device__ __forceinline__ void stage(const _Float16* __restrict__ A,
                                             const _Float16* __restrict__ W, int K,
                                             int kk, ldsh* lA, ldsh* lB, int tid) {
#pragma unroll
  for (int i = 0; i < 2; ++i) {
    const int idx = tid + i * 128;       // 0..255
    const int row = idx >> 2;            // 0..63
    const int ch  = (idx & 3) * 8;       // half-offset within 32-half row
    cp16(A + (size_t)row * K + kk + ch, lA + row * 32 + ch);
    cp16(W + (size_t)row * K + kk + ch, lB + row * 32 + ch);
  }
}

// ---------------------------------------------------------------------------
// Block-level 64x64 GEMM: C = A[64xK] * W[64xK]^T, f32 accum.
// 4 waves; wave w computes rows [16w,16w+16) x 64 cols (4 WMMA accumulators).
// Double-buffered LDS tiles staged with async copies.
// sm: [2 buffers][A 64x32 | B 64x32] = 16KB total.
// ---------------------------------------------------------------------------
static __device__ __forceinline__ void
block_gemm_64x64(const _Float16* __restrict__ A, const _Float16* __restrict__ W,
                 int K, ldsh* sm, int tid, v8f acc[4]) {
  const int wave = tid >> 5, lane = tid & 31;
  const int mrow = lane & 15, khalf = lane >> 4;
  const int nl = lane & 15, kb2 = khalf * 16;

  // buffer o: A at sm + o, B at sm + 2048 + o, with o in {0, 4096} (halves)
  stage(A, W, K, 0, sm, sm + 2048, tid);
  int buf = 0;
  for (int kk = 0; kk < K; kk += 32) {
    const bool more = (kk + 32) < K;
    const int o  = buf << 12;            // buf * 4096
    const int on = 4096 - o;             // other buffer
    if (more) stage(A, W, K, kk + 32, sm + on, sm + 2048 + on, tid);
    if (more) wait_async_le4(); else wait_async_0();
    __syncthreads();

    const ldsh* la = sm + o + (wave * 16 + mrow) * 32 + khalf * 16;
    const v16h af = ld16c_l(la);
#pragma unroll
    for (int n = 0; n < 4; ++n) {
      const ldsh* lb = sm + o + 2048 + (n * 16 + nl) * 32 + kb2;
      acc[n] = wmma_f16(af, ld16c_l(lb), acc[n]);
    }
    __syncthreads();
    buf ^= 1;
  }
}

// ---------------------------------------------------------------------------
// 1) fp32 -> fp16 conversion (8 elements / thread)
// ---------------------------------------------------------------------------
__global__ void cvt_f32_f16_v8(const float* __restrict__ src,
                               _Float16* __restrict__ dst, int n8) {
  const int i = blockIdx.x * blockDim.x + threadIdx.x;
  if (i < n8) {
    const float4* s = (const float4*)src + (size_t)i * 2;
    const float4 a = s[0], b = s[1];
    v8h o;
    o[0] = (_Float16)a.x; o[1] = (_Float16)a.y; o[2] = (_Float16)a.z; o[3] = (_Float16)a.w;
    o[4] = (_Float16)b.x; o[5] = (_Float16)b.y; o[6] = (_Float16)b.z; o[7] = (_Float16)b.w;
    ((v8h*)dst)[i] = o;
  }
}

// ---------------------------------------------------------------------------
// 2) QKV projection: qkv = Xh @ Wqkv^T + b.  Scatter Q,K -> [B,H,L,C],
//    V -> transposed [B,H,C,L].
// ---------------------------------------------------------------------------
__global__ void __launch_bounds__(128)
qkv_gemm(const _Float16* __restrict__ Xh, const _Float16* __restrict__ Wh,
         const float* __restrict__ bias, _Float16* __restrict__ Qh,
         _Float16* __restrict__ Kh, _Float16* __restrict__ VT) {
  __shared__ __align__(16) _Float16 smem[4 * 2048];
  ldsh* sm = (ldsh*)smem;
  const int tid = threadIdx.x;
  const int mt = blockIdx.x & 63;    // 64 M tiles (B*L/64)
  const int nt = blockIdx.x >> 6;    // 48 N tiles (3D/64)

  v8f acc[4];
#pragma unroll
  for (int n = 0; n < 4; ++n) acc[n] = (v8f){};
  block_gemm_64x64(Xh + (size_t)mt * 64 * D_, Wh + (size_t)nt * 64 * D_, D_,
                   sm, tid, acc);

  const int wave = tid >> 5, lane = tid & 31;
  const int nl = lane & 15, half = lane >> 4;
#pragma unroll
  for (int n = 0; n < 4; ++n) {
    const int gn = nt * 64 + n * 16 + nl;
    const float bv = bias[gn];
#pragma unroll
    for (int r = 0; r < 8; ++r) {
      const int gm = mt * 64 + wave * 16 + r + half * 8;  // [0, B*L)
      const int bb = gm >> 11;
      const int l  = gm & (L_ - 1);
      const float v = acc[n][r] + bv;
      if (gn < D_) {
        const int f = gn, hh = f >> 6, c = f & 63;
        Qh[(((size_t)(bb * H_ + hh) * L_) + l) * C_ + c] = (_Float16)v;
      } else if (gn < 2 * D_) {
        const int f = gn - D_, hh = f >> 6, c = f & 63;
        Kh[(((size_t)(bb * H_ + hh) * L_) + l) * C_ + c] = (_Float16)v;
      } else {
        const int f = gn - 2 * D_, hh = f >> 6, c = f & 63;
        VT[((size_t)(bb * H_ + hh) * C_ + c) * L_ + l] = (_Float16)v;
      }
    }
  }
}

// ---------------------------------------------------------------------------
// 3) Flash attention: one wave per (b,h, 16-query tile); streams 32 keys per
//    iteration; online softmax; O accumulated over C=64 as 4 WMMA tiles.
// ---------------------------------------------------------------------------
__global__ void __launch_bounds__(128)
flash_attn(const _Float16* __restrict__ Qh, const _Float16* __restrict__ Kh,
           const _Float16* __restrict__ VT, _Float16* __restrict__ Hh) {
  __shared__ __align__(16) _Float16 lp[4][16 * 32];   // P staging, 1KB/wave

  const int wave = threadIdx.x >> 5, lane = threadIdx.x & 31;
  const int g  = blockIdx.x * 4 + wave;   // 0..4095
  const int qt = g & 127;                 // 128 query tiles per (b,h)
  const int bh = g >> 7;                  // 0..31

  const _Float16* Qp = Qh + ((size_t)bh * L_ + qt * 16) * C_;
  const _Float16* Kp = Kh + (size_t)bh * L_ * C_;
  const _Float16* Vp = VT + (size_t)bh * C_ * L_;   // [C][L]

  const int mrow = lane & 15, khalf = lane >> 4;
  const int nl = lane & 15, kb2 = khalf * 16;

  const _Float16* qrow = Qp + mrow * C_ + khalf * 16;
  const v16h qa0 = ld16c(qrow);
  const v16h qa1 = ld16c(qrow + 32);

  v8f o0 = {}, o1 = {}, o2 = {}, o3 = {};
  float mst[8], lst[8];
#pragma unroll
  for (int r = 0; r < 8; ++r) { mst[r] = -1e30f; lst[r] = 0.0f; }

  ldsh* lpw = (ldsh*)&lp[wave][0];

  for (int jj = 0; jj < L_; jj += 32) {
    // S[16x32] = Q[16x64] . K[32x64]^T  (4 WMMAs)
    const _Float16* k0 = Kp + (size_t)(jj + nl) * C_ + kb2;
    const _Float16* k1 = Kp + (size_t)(jj + 16 + nl) * C_ + kb2;
    v8f s0 = {}, s1 = {};
    s0 = wmma_f16(qa0, ld16c(k0),      s0);
    s0 = wmma_f16(qa1, ld16c(k0 + 32), s0);
    s1 = wmma_f16(qa0, ld16c(k1),      s1);
    s1 = wmma_f16(qa1, ld16c(k1 + 32), s1);

    // online softmax; rows live across a 16-lane half -> xor 1,2,4,8
    float p0[8], p1[8];
#pragma unroll
    for (int r = 0; r < 8; ++r) {
      const float a = s0[r] * SCALE_;
      const float b = s1[r] * SCALE_;
      float mx = fmaxf(a, b);
      mx = fmaxf(mx, __shfl_xor(mx, 1, 32));
      mx = fmaxf(mx, __shfl_xor(mx, 2, 32));
      mx = fmaxf(mx, __shfl_xor(mx, 4, 32));
      mx = fmaxf(mx, __shfl_xor(mx, 8, 32));
      const float mn    = fmaxf(mst[r], mx);
      const float alpha = __expf(mst[r] - mn);
      mst[r] = mn;
      const float e0 = __expf(a - mn);
      const float e1 = __expf(b - mn);
      p0[r] = e0; p1[r] = e1;
      float rs = e0 + e1;
      rs += __shfl_xor(rs, 1, 32);
      rs += __shfl_xor(rs, 2, 32);
      rs += __shfl_xor(rs, 4, 32);
      rs += __shfl_xor(rs, 8, 32);
      lst[r] = lst[r] * alpha + rs;
      o0[r] *= alpha; o1[r] *= alpha; o2[r] *= alpha; o3[r] *= alpha;
    }

    // re-layout P (C-layout f32) -> A-fragment (f16) through LDS
#pragma unroll
    for (int r = 0; r < 8; ++r) {
      const int m = r + khalf * 8;
      lpw[m * 32 + nl]      = (_Float16)p0[r];
      lpw[m * 32 + 16 + nl] = (_Float16)p1[r];
    }
    __builtin_amdgcn_wave_barrier();
    const v16h pa = ld16c_l(lpw + mrow * 32 + khalf * 16);

    // O[16x64] += P[16x32] . V[32x64]  (4 WMMAs, V^T contiguous)
    const _Float16* vb = Vp + (size_t)nl * L_ + jj + kb2;
    o0 = wmma_f16(pa, ld16c(vb),           o0);
    o1 = wmma_f16(pa, ld16c(vb + 16 * L_), o1);
    o2 = wmma_f16(pa, ld16c(vb + 32 * L_), o2);
    o3 = wmma_f16(pa, ld16c(vb + 48 * L_), o3);
  }

  const int b = bh >> 4, h = bh & 15;
#pragma unroll
  for (int r = 0; r < 8; ++r) {
    const int m = r + khalf * 8;
    const size_t row = ((size_t)(b * L_ + qt * 16 + m)) * D_ + h * C_;
    const float inv = 1.0f / lst[r];
    Hh[row + nl]      = (_Float16)(o0[r] * inv);
    Hh[row + 16 + nl] = (_Float16)(o1[r] * inv);
    Hh[row + 32 + nl] = (_Float16)(o2[r] * inv);
    Hh[row + 48 + nl] = (_Float16)(o3[r] * inv);
  }
}

// ---------------------------------------------------------------------------
// 4) Output projection + bias + residual:  out = Hh @ Wo^T + b_o + x  (f32)
// ---------------------------------------------------------------------------
__global__ void __launch_bounds__(128)
out_gemm(const _Float16* __restrict__ Hh, const _Float16* __restrict__ Woh,
         const float* __restrict__ b_o, const float* __restrict__ x,
         float* __restrict__ out) {
  __shared__ __align__(16) _Float16 smem[4 * 2048];
  ldsh* sm = (ldsh*)smem;
  const int tid = threadIdx.x;
  const int mt = blockIdx.x & 63;    // 64 M tiles
  const int nt = blockIdx.x >> 6;    // 16 N tiles

  v8f acc[4];
#pragma unroll
  for (int n = 0; n < 4; ++n) acc[n] = (v8f){};
  block_gemm_64x64(Hh + (size_t)mt * 64 * D_, Woh + (size_t)nt * 64 * D_, D_,
                   sm, tid, acc);

  const int wave = tid >> 5, lane = tid & 31;
  const int nl = lane & 15, half = lane >> 4;
#pragma unroll
  for (int n = 0; n < 4; ++n) {
    const int gn = nt * 64 + n * 16 + nl;
    const float bv = b_o[gn];
#pragma unroll
    for (int r = 0; r < 8; ++r) {
      const int gm = mt * 64 + wave * 16 + r + half * 8;
      const size_t idx = (size_t)gm * D_ + gn;
      out[idx] = acc[n][r] + bv + x[idx];
    }
  }
}

// ---------------------------------------------------------------------------
// Host launcher
// ---------------------------------------------------------------------------
extern "C" void kernel_launch(void* const* d_in, const int* in_sizes, int n_in,
                              void* d_out, int out_size, void* d_ws, size_t ws_size,
                              hipStream_t stream) {
  (void)in_sizes; (void)n_in; (void)out_size; (void)ws_size;

  const float* x     = (const float*)d_in[0];
  const float* w_qkv = (const float*)d_in[1];
  const float* b_qkv = (const float*)d_in[2];
  const float* w_o   = (const float*)d_in[3];
  const float* b_o   = (const float*)d_in[4];
  float* out = (float*)d_out;

  const size_t nX    = (size_t)B_ * L_ * D_;      // 4,194,304
  const size_t nWqkv = (size_t)3 * D_ * D_;       // 3,145,728
  const size_t nWo   = (size_t)D_ * D_;           // 1,048,576
  const size_t nHead = (size_t)B_ * H_ * L_ * C_; // 4,194,304

  char* ws = (char*)d_ws;
  size_t off = 0;
  auto carve = [&](size_t bytes) {
    char* p = ws + off;
    off += (bytes + 255) & ~(size_t)255;
    return p;
  };
  _Float16* Xh    = (_Float16*)carve(nX * 2);
  _Float16* Wqkvh = (_Float16*)carve(nWqkv * 2);
  _Float16* Woh   = (_Float16*)carve(nWo * 2);
  _Float16* Qh    = (_Float16*)carve(nHead * 2);
  _Float16* Kh    = (_Float16*)carve(nHead * 2);
  _Float16* VT    = (_Float16*)carve(nHead * 2);
  _Float16* Hh    = (_Float16*)carve(nX * 2);

  // 1) convert inputs to f16 (8 elems/thread)
  cvt_f32_f16_v8<<<(int)((nX / 8    + 255) / 256), 256, 0, stream>>>(x,     Xh,    (int)(nX / 8));
  cvt_f32_f16_v8<<<(int)((nWqkv / 8 + 255) / 256), 256, 0, stream>>>(w_qkv, Wqkvh, (int)(nWqkv / 8));
  cvt_f32_f16_v8<<<(int)((nWo / 8   + 255) / 256), 256, 0, stream>>>(w_o,   Woh,   (int)(nWo / 8));

  // 2) QKV projection: 64 x 48 block tiles of 64x64
  qkv_gemm<<<64 * 48, 128, 0, stream>>>(Xh, Wqkvh, b_qkv, Qh, Kh, VT);

  // 3) flash attention: B*H*(L/16) = 4096 waves, 4 waves/block
  flash_attn<<<4096 / 4, 128, 0, stream>>>(Qh, Kh, VT, Hh);

  // 4) output projection + residual: 64 x 16 block tiles of 64x64
  out_gemm<<<64 * 16, 128, 0, stream>>>(Hh, Woh, b_o, x, out);
}
